// SpatialAttention_54065048322479
// MI455X (gfx1250) — matrix operand order repaired and checked by reading, not compiled, not run on hardware
//
#include <hip/hip_runtime.h>
#include <hip/hip_bf16.h>

// ---------------------------------------------------------------------------
// SpatialAttention for MI455X (gfx1250, wave32).
//   L = flat @ W1^T, R = flat @ W2^T           -> v_wmma_f32_16x16x32_bf16
//   scores[b,i,j] = sum_m sigmoid(L[i,m]+R[j,m]) = 256 + 0.5*sum_m tanh(.5*(L+R))
//                                               -> v_tanh_f32 (gfx1250 TRANS)
//   softmax_j(scores + Vs)                      -> wave32 shfl_xor reductions
// ---------------------------------------------------------------------------

typedef __attribute__((ext_vector_type(16))) __bf16 v16bf;
typedef __attribute__((ext_vector_type(8)))  float  v8f;

#define NN   512      // nodes (rows of W, softmax width)
#define DD   1024     // channels*timesteps
#define BN   1024     // B * N  (rows of flat)

// ---------------------------------------------------------------------------
// Kernel 1: one wave computes one 16x16 tile of BOTH L and R (shares the A
// fragment). Fragment k order used here (contiguous, same for A and B):
//   A (16x32 bf16): lane l -> M = l%16; VGPR v element pair covers
//       K = (l>=16?16:0) + 2v + {0,1}
//   B (32x16 bf16): lane l -> N = l%16; VGPR v element pair covers
//       K = (l>=16?16:0) + 2v + {0,1}
// Both X (row-major [BN,DD]) and W (row-major [NN,DD]) are K-contiguous, so
// each fragment element pair is one aligned float2 load + cvt to bf16.
// ---------------------------------------------------------------------------
__global__ __launch_bounds__(256)
void proj_gemm_wmma(const float* __restrict__ X,   // [BN, DD]
                    const float* __restrict__ W1,  // [NN, DD]
                    const float* __restrict__ W2,  // [NN, DD]
                    float* __restrict__ L,         // [BN, NN]
                    float* __restrict__ R)         // [BN, NN]
{
    const int wave = blockIdx.x * (blockDim.x >> 5) + (threadIdx.x >> 5);
    const int lane = threadIdx.x & 31;
    const int trow = wave >> 5;           // 0..63 : 16-row tile of flat
    const int tcol = wave & 31;           // 0..31 : 16-col tile of W rows
    const int row0 = trow << 4;
    const int col0 = tcol << 4;

    const int mn   = lane & 15;           // M for A, N for B
    const int half = lane >> 4;

    const float* Arow  = X  + (row0 + mn) * DD;
    const float* B1row = W1 + (col0 + mn) * DD;
    const float* B2row = W2 + (col0 + mn) * DD;

    v8f c1 = {};
    v8f c2 = {};

    for (int k = 0; k < DD; k += 32) {
        v16bf a, b1, b2;
#pragma unroll
        for (int v = 0; v < 8; ++v) {
            const int ka = half * 16 + v * 2;
            const int kb = half * 16 + v * 2;
            float2 fa = *(const float2*)(Arow  + k + ka);
            float2 f1 = *(const float2*)(B1row + k + kb);
            float2 f2 = *(const float2*)(B2row + k + kb);
            a [2 * v] = (__bf16)fa.x;  a [2 * v + 1] = (__bf16)fa.y;
            b1[2 * v] = (__bf16)f1.x;  b1[2 * v + 1] = (__bf16)f1.y;
            b2[2 * v] = (__bf16)f2.x;  b2[2 * v + 1] = (__bf16)f2.y;
        }
        c1 = __builtin_amdgcn_wmma_f32_16x16x32_bf16(false, a, false, b1,
                                                     (short)0, c1, false, false);
        c2 = __builtin_amdgcn_wmma_f32_16x16x32_bf16(false, a, false, b2,
                                                     (short)0, c2, false, false);
    }

    // C/D layout: VGPR r -> M = r + 8*half, N = lane%16
#pragma unroll
    for (int r = 0; r < 8; ++r) {
        const int gi = (row0 + r + half * 8) * NN + col0 + mn;
        L[gi] = c1[r];
        R[gi] = c2[r];
    }
}

// ---------------------------------------------------------------------------
// Kernel 2: one block = (batch b, 16 rows i). Stages 0.5*L[i-tile] in LDS
// (broadcast reads), each thread owns R rows j=t and j=t+256, accumulates
//   acc[i] += tanh(0.5*L[i,m] + 0.5*R[j,m])      via v_tanh_f32 (1 TRANS/term)
// then scores = 256 + 0.5*acc + Vs, and a wave32 softmax per row.
// ---------------------------------------------------------------------------
__global__ __launch_bounds__(256)
void attn_tanh_softmax(const float* __restrict__ L,
                       const float* __restrict__ R,
                       const float* __restrict__ Vs,   // [NN, NN]
                       float* __restrict__ out)        // [2, NN, NN]
{
    __shared__ float smem[16 * NN];       // 32 KB: 0.5*L tile, then scores

    const int b  = blockIdx.x >> 5;
    const int i0 = (blockIdx.x & 31) << 4;
    const int t  = threadIdx.x;

    // ---- stage pre-halved L tile (contiguous 8192 floats, coalesced) ----
    const float* Lbase = L + (b * NN + i0) * NN;
#pragma unroll
    for (int kk = 0; kk < 32; ++kk) {
        const int idx = t + kk * 256;
        smem[idx] = 0.5f * Lbase[idx];
    }
    __syncthreads();

    const int ja = t;
    const int jb = t + 256;
    const float* Ra = R + (b * NN + ja) * NN;
    const float* Rb = R + (b * NN + jb) * NN;

    float accA[16], accB[16];
#pragma unroll
    for (int i = 0; i < 16; ++i) { accA[i] = 0.0f; accB[i] = 0.0f; }

    for (int m = 0; m < NN; m += 4) {
        const float4 r4a = *(const float4*)(Ra + m);
        const float4 r4b = *(const float4*)(Rb + m);
#pragma unroll
        for (int s = 0; s < 4; ++s) {
            const float ra = 0.5f * ((const float*)&r4a)[s];
            const float rb = 0.5f * ((const float*)&r4b)[s];
            float ta[16], tb[16];
#pragma unroll
            for (int i = 0; i < 16; ++i) {
                const float hl = smem[i * NN + m + s];   // LDS broadcast
                const float xa = hl + ra;
                const float xb = hl + rb;
                asm("v_tanh_f32 %0, %1" : "=v"(ta[i]) : "v"(xa));
                asm("v_tanh_f32 %0, %1" : "=v"(tb[i]) : "v"(xb));
            }
            // adds batched after the tanh burst: separates TRANS results
            // from their consumers (co-execution hazard, ISA 7.4)
#pragma unroll
            for (int i = 0; i < 16; ++i) { accA[i] += ta[i]; accB[i] += tb[i]; }
        }
    }
    __syncthreads();   // everyone done reading the L tile

    // ---- scores = 256 + 0.5*sum(tanh) + Vs, back into LDS ----
#pragma unroll
    for (int i = 0; i < 16; ++i) {
        const int vi = (i0 + i) * NN;
        smem[i * NN + ja] = 256.0f + 0.5f * accA[i] + Vs[vi + ja];
        smem[i * NN + jb] = 256.0f + 0.5f * accB[i] + Vs[vi + jb];
    }
    __syncthreads();

    // ---- softmax over j: wave w handles rows 2w, 2w+1 (wave32) ----
    const int wv   = t >> 5;
    const int lane = t & 31;
    const float LOG2E = 1.4426950408889634f;

#pragma unroll
    for (int rr = 0; rr < 2; ++rr) {
        const int i = wv * 2 + rr;
        const float* srow = smem + i * NN;

        float mx = -3.0e38f;
#pragma unroll
        for (int kk = 0; kk < 16; ++kk) mx = fmaxf(mx, srow[lane + kk * 32]);
#pragma unroll
        for (int off = 16; off > 0; off >>= 1)
            mx = fmaxf(mx, __shfl_xor(mx, off, 32));

        float e[16];
        float sum = 0.0f;
#pragma unroll
        for (int kk = 0; kk < 16; ++kk) {
            e[kk] = __builtin_amdgcn_exp2f((srow[lane + kk * 32] - mx) * LOG2E);
            sum += e[kk];
        }
#pragma unroll
        for (int off = 16; off > 0; off >>= 1)
            sum += __shfl_xor(sum, off, 32);

        const float inv = 1.0f / sum;
        float* orow = out + (b * NN + i0 + i) * NN;
#pragma unroll
        for (int kk = 0; kk < 16; ++kk)
            orow[lane + kk * 32] = e[kk] * inv;
    }
}

// ---------------------------------------------------------------------------
extern "C" void kernel_launch(void* const* d_in, const int* in_sizes, int n_in,
                              void* d_out, int out_size, void* d_ws, size_t ws_size,
                              hipStream_t stream)
{
    const float* x  = (const float*)d_in[0];   // [2,512,64,16] == flat [1024,1024]
    const float* W1 = (const float*)d_in[1];   // [512,1024]
    const float* W2 = (const float*)d_in[2];   // [512,1024]
    const float* Vs = (const float*)d_in[3];   // [512,512]
    float* out = (float*)d_out;                // [2,512,512]

    float* L = (float*)d_ws;                   // 2 MB
    float* R = L + 2 * NN * NN;                // 2 MB  (needs ws_size >= 4 MB)

    // 2048 output tiles (64 row-tiles x 32 col-tiles), 1 wave each, 8 waves/block
    proj_gemm_wmma<<<256, 256, 0, stream>>>(x, W1, W2, L, R);
    // 64 blocks: (b, 16-row tile); each does sigmoid-sum + Vs + softmax
    attn_tanh_softmax<<<64, 256, 0, stream>>>(L, R, Vs, out);
}
